// APPNPNet_5643587027246
// MI455X (gfx1250) — matrix operand order repaired and checked by reading, not compiled, not run on hardware
//
#include <hip/hip_runtime.h>
#include <hip/hip_bf16.h>

typedef __attribute__((ext_vector_type(16))) _Float16 v16h;
typedef __attribute__((ext_vector_type(8)))  float    v8f;

#define ALPHA   0.15f
#define BETA    0.85f   // 1 - alpha
#define HID     64
#define K_HOPS  10

// ---------------------------------------------------------------------------
// deg[i] = 1 (self loop)
__global__ void k_init_deg(float* __restrict__ deg, int n) {
    int i = blockIdx.x * blockDim.x + threadIdx.x;
    if (i < n) deg[i] = 1.0f;
}

// deg[dst] += 1 per edge
__global__ void k_count_deg(const int* __restrict__ edst, float* __restrict__ deg, int E) {
    int e = blockIdx.x * blockDim.x + threadIdx.x;
    if (e < E) atomicAdd(&deg[edst[e]], 1.0f);
}

// dinv = rsqrt(deg)   (deg >= 1 always, matches reference's guarded rsqrt)
__global__ void k_dinv(float* __restrict__ deg, int n) {
    int i = blockIdx.x * blockDim.x + threadIdx.x;
    if (i < n) deg[i] = __frsqrt_rn(deg[i]);
}

// zero a float buffer (vectorized)
__global__ void k_zero(float4* __restrict__ p, long n4) {
    long i = (long)blockIdx.x * blockDim.x + threadIdx.x;
    if (i < n4) p[i] = make_float4(0.f, 0.f, 0.f, 0.f);
}

// hidden[row, :] += W1[col, :]  -- 32 lanes per nnz, 2 channels each
__global__ void k_attr_scatter(const int* __restrict__ arow, const int* __restrict__ acol,
                               const float* __restrict__ W1, float* __restrict__ hidden,
                               int nnz) {
    long t = (long)blockIdx.x * blockDim.x + threadIdx.x;
    int a = (int)(t >> 5);
    int lane = (int)(t & 31);
    if (a >= nnz) return;
    int r = arow[a], c = acol[a];
    const float* w = W1 + (long)c * HID;
    float* h = hidden + (long)r * HID;
    atomicAdd(&h[lane],      w[lane]);
    atomicAdd(&h[lane + 32], w[lane + 32]);
}

// ---------------------------------------------------------------------------
// relu(hidden[n,64]) @ W2[64,64] -> outA[n,64] (aliases hidden, per-tile safe)
// and outH[n,64]. One wave per 16x64 output tile, v_wmma_f32_16x16x32_f16.
__global__ void k_gemm_wmma(const float* __restrict__ hidden, const float* __restrict__ W2,
                            float* __restrict__ outA, float* __restrict__ outH, int n) {
    int lane = threadIdx.x & 31;
    int half = lane >> 4;       // 0 or 1
    int l16  = lane & 15;
    int waveId = (int)((blockIdx.x * blockDim.x + threadIdx.x) >> 5);
    int nWaves = (int)((gridDim.x * blockDim.x) >> 5);
    int nTiles = n >> 4;        // n % 16 == 0 for this problem

    // ---- load B fragments once per wave: 4 N-tiles x 2 K-blocks ----
    // B frag k order used here (contiguous, same as A): lane half h covers K = 16h+2i+j, N = l16
    v16h bfrag[4][2];
#pragma unroll
    for (int nt = 0; nt < 4; ++nt)
#pragma unroll
        for (int kb = 0; kb < 2; ++kb) {
            v16h b;
#pragma unroll
            for (int i = 0; i < 8; ++i)
#pragma unroll
                for (int j = 0; j < 2; ++j) {
                    int K = 32 * kb + 16 * half + 2 * i + j;
                    b[2 * i + j] = (_Float16)W2[K * HID + nt * 16 + l16];
                }
            bfrag[nt][kb] = b;
        }

    for (int mt = waveId; mt < nTiles; mt += nWaves) {
        int m0 = mt << 4;
        // ---- load A fragments: relu(hidden) rows m0..m0+15, f32 -> f16 ----
        // A frag k order used here (contiguous, same as B): lane half h covers K = 16h+0..15
        v16h afrag[2];
#pragma unroll
        for (int kb = 0; kb < 2; ++kb) {
            const float* rp = hidden + (long)(m0 + l16) * HID + 32 * kb + 16 * half;
            float4 p0 = *(const float4*)(rp);
            float4 p1 = *(const float4*)(rp + 4);
            float4 p2 = *(const float4*)(rp + 8);
            float4 p3 = *(const float4*)(rp + 12);
            v16h a;
            a[0]  = (_Float16)fmaxf(p0.x, 0.f); a[1]  = (_Float16)fmaxf(p0.y, 0.f);
            a[2]  = (_Float16)fmaxf(p0.z, 0.f); a[3]  = (_Float16)fmaxf(p0.w, 0.f);
            a[4]  = (_Float16)fmaxf(p1.x, 0.f); a[5]  = (_Float16)fmaxf(p1.y, 0.f);
            a[6]  = (_Float16)fmaxf(p1.z, 0.f); a[7]  = (_Float16)fmaxf(p1.w, 0.f);
            a[8]  = (_Float16)fmaxf(p2.x, 0.f); a[9]  = (_Float16)fmaxf(p2.y, 0.f);
            a[10] = (_Float16)fmaxf(p2.z, 0.f); a[11] = (_Float16)fmaxf(p2.w, 0.f);
            a[12] = (_Float16)fmaxf(p3.x, 0.f); a[13] = (_Float16)fmaxf(p3.y, 0.f);
            a[14] = (_Float16)fmaxf(p3.z, 0.f); a[15] = (_Float16)fmaxf(p3.w, 0.f);
            afrag[kb] = a;
        }

        // Per-tile base pointers: every store below is base + compile-time
        // constant offset (r*HID floats = r*256B, nt*16 floats = nt*64B),
        // so they lower to global_store_b32 with immediate offsets.
        float* baseA = outA + (long)(m0 + 8 * half) * HID + l16;
        float* baseH = outH + (long)(m0 + 8 * half) * HID + l16;

        // ---- 4 N-tiles, K=64 via two chained WMMAs each ----
#pragma unroll
        for (int nt = 0; nt < 4; ++nt) {
            v8f c = {};
            c = __builtin_amdgcn_wmma_f32_16x16x32_f16(false, afrag[0], false, bfrag[nt][0],
                                                       (short)0, c, false, false);
            c = __builtin_amdgcn_wmma_f32_16x16x32_f16(false, afrag[1], false, bfrag[nt][1],
                                                       (short)0, c, false, false);
            // C/D layout: VGPR r -> M = r + 8*half, N = l16
#pragma unroll
            for (int r = 0; r < 8; ++r) {
                float v = c[r];
                baseA[nt * 16 + r * HID] = v;
                baseH[nt * 16 + r * HID] = v;
            }
        }
    }
}

// ---------------------------------------------------------------------------
// next = beta * dinv[i]^2 * cur + alpha * h   (self-loop + teleport; full overwrite)
__global__ void k_prop_init(const float* __restrict__ cur, const float* __restrict__ h,
                            const float* __restrict__ dinv, float* __restrict__ next, int n) {
    long t = (long)blockIdx.x * blockDim.x + threadIdx.x;   // one float4 per thread
    long n4 = (long)n * (HID / 4);
    if (t >= n4) return;
    int node = (int)(t >> 4);
    float di = dinv[node];
    float w = BETA * di * di;
    float4 x = ((const float4*)cur)[t];
    float4 hh = ((const float4*)h)[t];
    float4 o;
    o.x = w * x.x + ALPHA * hh.x;
    o.y = w * x.y + ALPHA * hh.y;
    o.z = w * x.z + ALPHA * hh.z;
    o.w = w * x.w + ALPHA * hh.w;
    ((float4*)next)[t] = o;
}

// next[dst] += beta * dinv[src]*dinv[dst] * cur[src]   -- 32 lanes/edge, 2 ch each
__global__ void k_prop_edges(const int* __restrict__ esrc, const int* __restrict__ edst,
                             const float* __restrict__ dinv, const float* __restrict__ cur,
                             float* __restrict__ next, int E) {
    long t = (long)blockIdx.x * blockDim.x + threadIdx.x;
    int e = (int)(t >> 5);
    int lane = (int)(t & 31);
    if (e >= E) return;
    int s = esrc[e], d = edst[e];
    float w = BETA * dinv[s] * dinv[d];
    const float* xs = cur + (long)s * HID;
    float* xd = next + (long)d * HID;
    atomicAdd(&xd[lane],      w * xs[lane]);
    atomicAdd(&xd[lane + 32], w * xs[lane + 32]);
}

// ---------------------------------------------------------------------------
// row-wise log_softmax over 64 classes: one wave32 per row, shfl reductions
__global__ void k_logsoftmax(const float* __restrict__ x, float* __restrict__ out, int n) {
    long t = (long)blockIdx.x * blockDim.x + threadIdx.x;
    int row = (int)(t >> 5);
    int lane = (int)(t & 31);
    if (row >= n) return;
    const float* r = x + (long)row * HID;
    float a = r[lane], b = r[lane + 32];
    float m = fmaxf(a, b);
#pragma unroll
    for (int off = 16; off > 0; off >>= 1) m = fmaxf(m, __shfl_xor(m, off, 32));
    float s = __expf(a - m) + __expf(b - m);
#pragma unroll
    for (int off = 16; off > 0; off >>= 1) s += __shfl_xor(s, off, 32);
    float lse = m + __logf(s);
    out[(long)row * HID + lane]      = a - lse;
    out[(long)row * HID + lane + 32] = b - lse;
}

// ---------------------------------------------------------------------------
extern "C" void kernel_launch(void* const* d_in, const int* in_sizes, int n_in,
                              void* d_out, int out_size, void* d_ws, size_t ws_size,
                              hipStream_t stream) {
    const int*   attr = (const int*)d_in[0];     // [2, nnz] : row0 = node, row1 = feat
    const int*   edge = (const int*)d_in[1];     // [2, E]   : row0 = src,  row1 = dst
    const float* W1   = (const float*)d_in[4];   // [8192, 64]
    const float* W2   = (const float*)d_in[5];   // [64, 64]

    const int nnz = in_sizes[0] / 2;
    const int E   = in_sizes[1] / 2;
    const int n   = out_size / HID;              // 100000

    const int* arow = attr;
    const int* acol = attr + nnz;
    const int* esrc = edge;
    const int* edst = edge + E;

    // workspace layout: dinv | bufA (hidden -> x ping) | bufH (teleport term)
    float* dinv = (float*)d_ws;
    size_t off  = ((size_t)n * sizeof(float) + 255) & ~(size_t)255;
    float* bufA = (float*)((char*)d_ws + off);
    float* bufH = bufA + (size_t)n * HID;
    float* xOut = (float*)d_out;                 // ping-pong partner

    const int T = 256;

    // degrees -> dinv
    k_init_deg<<<(n + T - 1) / T, T, 0, stream>>>(dinv, n);
    k_count_deg<<<(E + T - 1) / T, T, 0, stream>>>(edst, dinv, E);
    k_dinv<<<(n + T - 1) / T, T, 0, stream>>>(dinv, n);

    // sparse feature SpMM: hidden = A_attr @ W1  (binary values)
    long n4 = (long)n * (HID / 4);
    k_zero<<<(unsigned)((n4 + T - 1) / T), T, 0, stream>>>((float4*)bufA, n4);
    long tScat = (long)nnz * 32;
    k_attr_scatter<<<(unsigned)((tScat + T - 1) / T), T, 0, stream>>>(arow, acol, W1, bufA, nnz);

    // dense GEMM via WMMA: bufA (in place) and bufH get relu(hidden) @ W2
    k_gemm_wmma<<<256, T, 0, stream>>>(bufA, W2, bufA, bufH, n);

    // APPNP: 10 hops ping-ponging bufA <-> d_out; hop parity lands final in bufA
    float* cur = bufA;
    float* nxt = xOut;
    long tEdge = (long)E * 32;
    for (int k = 0; k < K_HOPS; ++k) {
        k_prop_init<<<(unsigned)((n4 + T - 1) / T), T, 0, stream>>>(cur, bufH, dinv, nxt, n);
        k_prop_edges<<<(unsigned)((tEdge + T - 1) / T), T, 0, stream>>>(esrc, edst, dinv, cur, nxt, E);
        float* tmp = cur; cur = nxt; nxt = tmp;
    }
    // after 10 hops cur == bufA

    long tRow = (long)n * 32;
    k_logsoftmax<<<(unsigned)((tRow + T - 1) / T), T, 0, stream>>>(cur, (float*)d_out, n);
}
